// MultiHeadAttention_40510131536016
// MI455X (gfx1250) — matrix operand order repaired and checked by reading, not compiled, not run on hardware
//
#include <hip/hip_runtime.h>
#include <stdint.h>

#define MODEL_DIM 1024
#define NUM_HEADS 16
#define HEAD_DIM  64
#define BATCH     4
#define SEQ       2048
#define MROWS     (BATCH * SEQ)   // 8192

typedef __attribute__((ext_vector_type(16))) __bf16    v16bf;
typedef __attribute__((ext_vector_type(8)))  float     v8f;
typedef __attribute__((ext_vector_type(4)))  uint32_t  v4u;
typedef __attribute__((ext_vector_type(8)))  int       v8i;
typedef __attribute__((ext_vector_type(4)))  int       v4i;

union Frag {
    v16bf    v;
    uint32_t u[8];
};

__device__ __forceinline__ uint16_t f2bf(float f) {
    uint32_t u = __float_as_uint(f);
    u += 0x7FFFu + ((u >> 16) & 1u);   // round-to-nearest-even
    return (uint16_t)(u >> 16);
}

__device__ __forceinline__ v8f wmma_bf16(const Frag& a, const Frag& b, v8f c) {
    return __builtin_amdgcn_wmma_f32_16x16x32_bf16(
        false, a.v, false, b.v, (short)0, c, false, false);
}

// Async 16-byte global->LDS copy (GVS mode: saddr base + per-lane voffset).
// Tracked on ASYNCcnt; pair with s_wait_asynccnt before the consuming barrier.
__device__ __forceinline__ void async_b128(uint32_t lds_addr, const void* base,
                                           uint32_t voff) {
    asm volatile("global_load_async_to_lds_b128 %0, %1, %2"
                 :: "v"(lds_addr), "v"(voff),
                    "s"((uint64_t)(uintptr_t)base)
                 : "memory");
}
__device__ __forceinline__ void wait_async0() {
    asm volatile("s_wait_asynccnt 0x0" ::: "memory");
}

#if __has_builtin(__builtin_amdgcn_tensor_load_to_lds)
#define HAVE_TDM 1
// 2D TDM load: nrows x rowlen bf16 tile, contiguous rows of `stride` elements.
// 6-arg builtin variant (clang-23 / therock-10.0 lane).
__device__ __forceinline__ void tdm_load_2d(uint32_t lds_addr, const void* gaddr,
                                            uint32_t rowlen, uint32_t nrows,
                                            uint32_t stride) {
    const uint64_t ga = (uint64_t)(uintptr_t)gaddr;
    v4u g0;
    g0[0] = 1u;                                            // count=1, user mode
    g0[1] = lds_addr;                                      // lds_addr (bytes)
    g0[2] = (uint32_t)ga;                                  // global_addr[31:0]
    g0[3] = (uint32_t)((ga >> 32) & 0x01FFFFFFu) | (2u << 30); // [56:32] | type=2
    v8i g1;
    g1[0] = (int)(1u << 16);                               // data_size=1 (2B)
    g1[1] = (int)((rowlen & 0xFFFFu) << 16);               // tensor_dim0 lo
    g1[2] = (int)(((rowlen >> 16) & 0xFFFFu) | ((nrows & 0xFFFFu) << 16));
    g1[3] = (int)(((nrows >> 16) & 0xFFFFu) | ((rowlen & 0xFFFFu) << 16)); // tile_dim0
    g1[4] = (int)(nrows & 0xFFFFu);                        // tile_dim1 (tile_dim2=0)
    g1[5] = (int)stride;                                   // tensor_dim0_stride lo
    g1[6] = 0;
    g1[7] = 0;
    const v4i z4 = {0, 0, 0, 0};
    const v8i z8 = {0, 0, 0, 0, 0, 0, 0, 0};
    __builtin_amdgcn_tensor_load_to_lds(g0, g1, z4, z4, z8, 0);
}
#else
#define HAVE_TDM 0
#endif

// ---------------------------------------------------------------------------
// fp32 -> bf16 elementwise convert (weights pre-pass)
// ---------------------------------------------------------------------------
__global__ __launch_bounds__(256)
void cvt_bf16_kernel(const float* __restrict__ in, uint16_t* __restrict__ out,
                     int n) {
    const int i = (blockIdx.x * 256 + threadIdx.x) * 4;
    if (i < n) {
        const float4 f = *(const float4*)(in + i);
        uint2 p;
        p.x = (uint32_t)f2bf(f.x) | ((uint32_t)f2bf(f.y) << 16);
        p.y = (uint32_t)f2bf(f.z) | ((uint32_t)f2bf(f.w) << 16);
        *(uint2*)(out + i) = p;
    }
}

// ---------------------------------------------------------------------------
// GEMM + bias:  C[m,n] = sum_k A[m,k] * W[n,k] + bias[n]   (W pre-converted bf16)
// MODE 0: A fp32 [8192,1024]; out bf16 head-split [B,H,S,hd], scaled.
// MODE 1: A bf16 [8192,1024]; out fp32 row-major [8192,1024].
// Block 256 thr = 8 waves (4 M x 2 N); block tile 128x128; wave tile 32x64.
// ---------------------------------------------------------------------------
template <int MODE>
__global__ __launch_bounds__(256)
void gemm_bias_kernel(const void*     __restrict__ Aptr,
                      const uint16_t* __restrict__ Wb,
                      const float*    __restrict__ bias,
                      void*           __restrict__ Out,
                      float scale)
{
    __shared__ __align__(16) uint16_t ldsA[128 * 32];
    __shared__ __align__(16) uint16_t ldsW[128 * 32];

    const int tid  = threadIdx.x;
    const int lane = tid & 31;
    const int wv   = tid >> 5;
    const int wm   = wv & 3;        // 4 M groups of 32 rows
    const int wn   = wv >> 2;       // 2 N groups of 64 cols
    const int lm   = lane & 15;
    const int hi   = lane >> 4;

    const int mBase = blockIdx.y * 128;
    const int nBase = blockIdx.x * 128;

    const v8f vzero = {0.f, 0.f, 0.f, 0.f, 0.f, 0.f, 0.f, 0.f};
    v8f acc[2][4];
#pragma unroll
    for (int rt = 0; rt < 2; ++rt)
#pragma unroll
        for (int ct = 0; ct < 4; ++ct) acc[rt][ct] = vzero;

    const float*    Af = (const float*)Aptr;
    const uint16_t* Ab = (const uint16_t*)Aptr;

    for (int kb = 0; kb < MODEL_DIM; kb += 32) {
        __syncthreads();   // previous iteration's fragment reads complete

        // ---- W tile: 128 rows x 32 k bf16 via async b128 (2 chunks/thread) ----
#pragma unroll
        for (int i = 0; i < 2; ++i) {
            const int c   = tid + 256 * i;      // 0..511 chunks of 16B
            const int row = c >> 2, seg = c & 3;
            const uint32_t voff = (uint32_t)((((nBase + row) * MODEL_DIM) + kb + seg * 8) * 2);
            async_b128((uint32_t)(uintptr_t)&ldsW[row * 32 + seg * 8], Wb, voff);
        }
        // ---- A tile: 128 rows x 32 k ----
        if (MODE == 0) {
            const int row = tid >> 1;
            const int kh  = (tid & 1) * 16;
            const float* src = Af + (size_t)(mBase + row) * MODEL_DIM + kb + kh;
            if (kb + 32 < MODEL_DIM) __builtin_prefetch(src + 32, 0, 0);
#pragma unroll
            for (int i = 0; i < 16; ++i)
                ldsA[row * 32 + kh + i] = f2bf(src[i]);
        } else {
#pragma unroll
            for (int i = 0; i < 2; ++i) {
                const int c   = tid + 256 * i;
                const int row = c >> 2, seg = c & 3;
                const uint32_t voff = (uint32_t)((((mBase + row) * MODEL_DIM) + kb + seg * 8) * 2);
                async_b128((uint32_t)(uintptr_t)&ldsA[row * 32 + seg * 8], Ab, voff);
            }
        }
        wait_async0();
        __syncthreads();

        // ---- fragments (ISA 7.12.2 layouts) ----
        Frag afr[2], bfr[4];
#pragma unroll
        for (int rt = 0; rt < 2; ++rt) {
            const uint32_t* rowp = (const uint32_t*)&ldsA[(wm * 32 + rt * 16 + lm) * 32];
#pragma unroll
            for (int v = 0; v < 8; ++v)
                afr[rt].u[v] = rowp[(v < 4 ? v : v + 4) + hi * 4];
        }
#pragma unroll
        for (int ct = 0; ct < 4; ++ct) {
            const uint32_t* rowp = (const uint32_t*)&ldsW[(wn * 64 + ct * 16 + lm) * 32];
#pragma unroll
            for (int v = 0; v < 8; ++v)
                bfr[ct].u[v] = rowp[(v < 4 ? v : v + 4) + hi * 4];
        }
#pragma unroll
        for (int rt = 0; rt < 2; ++rt)
#pragma unroll
            for (int ct = 0; ct < 4; ++ct)
                acc[rt][ct] = wmma_bf16(afr[rt], bfr[ct], acc[rt][ct]);
    }

    // ---- epilogue ----
#pragma unroll
    for (int rt = 0; rt < 2; ++rt) {
#pragma unroll
        for (int ct = 0; ct < 4; ++ct) {
            const int   col  = nBase + wn * 64 + ct * 16 + lm;
            const float bv   = bias[col];
            const int   row0 = mBase + wm * 32 + rt * 16 + hi * 8;
            if (MODE == 0) {
                uint16_t* dst = (uint16_t*)Out;
                const int h = col >> 6, d = col & 63;
#pragma unroll
                for (int r = 0; r < 8; ++r) {
                    const int row = row0 + r;
                    const int b = row >> 11, s = row & (SEQ - 1);
                    dst[((size_t)(b * NUM_HEADS + h) * SEQ + s) * HEAD_DIM + d] =
                        f2bf((acc[rt][ct][r] + bv) * scale);
                }
            } else {
                float* dst = (float*)Out;
#pragma unroll
                for (int r = 0; r < 8; ++r) {
                    const int row = row0 + r;
                    dst[(size_t)row * MODEL_DIM + col] = acc[rt][ct][r] + bv;
                }
            }
        }
    }
}

// ---------------------------------------------------------------------------
// Flash attention over bf16 Q/K/V in [B*H, S, hd] layout (1/sqrt(hd) in Q).
// Block 128 thr = 4 waves; each wave owns 16 query rows of a 64-row tile.
// K tile staged by the Tensor Data Mover (wave 0), V staged transposed.
// ---------------------------------------------------------------------------
__global__ __launch_bounds__(128)
void attention_kernel(const uint16_t* __restrict__ Qh,
                      const uint16_t* __restrict__ Kh,
                      const uint16_t* __restrict__ Vh,
                      uint16_t*       __restrict__ ctx)
{
    __shared__ __align__(16) uint16_t Kt[64 * HEAD_DIM];   // [key][d]
    __shared__ __align__(16) uint16_t Vt[HEAD_DIM * 64];   // [d][key]
    __shared__ __align__(16) uint16_t Pb[4][16 * 64];      // per-wave P tile

    const int tid  = threadIdx.x;
    const int lane = tid & 31;
    const int w    = tid >> 5;
    const int lm   = lane & 15;
    const int hi   = lane >> 4;

    const int    qt = blockIdx.x;
    const int    bh = blockIdx.y;
    const int    q0 = qt * 64;
    const size_t base = (size_t)bh * SEQ * HEAD_DIM;
    const uint16_t* Q = Qh + base;
    const uint16_t* K = Kh + base;
    const uint16_t* V = Vh + base;

    const v8f vzero = {0.f, 0.f, 0.f, 0.f, 0.f, 0.f, 0.f, 0.f};

    // Q fragments (2 k-steps of 32 over hd=64), direct from global
    Frag qfr[2];
    {
        const uint32_t* rowp = (const uint32_t*)(Q + (size_t)(q0 + w * 16 + lm) * HEAD_DIM);
#pragma unroll
        for (int kk = 0; kk < 2; ++kk)
#pragma unroll
            for (int v = 0; v < 8; ++v)
                qfr[kk].u[v] = rowp[kk * 16 + (v < 4 ? v : v + 4) + hi * 4];
    }

    float mrow[8], lrow[8];
    v8f   oacc[4];
#pragma unroll
    for (int r = 0; r < 8; ++r) { mrow[r] = -1e30f; lrow[r] = 0.f; }
#pragma unroll
    for (int n = 0; n < 4; ++n) oacc[n] = vzero;

    for (int t = 0; t <= qt; ++t) {
        const int kv0 = t * 64;

        __syncthreads();   // previous iteration's LDS reads complete

        // ---- K tile 64x64 bf16 via Tensor Data Mover (wave 0 issues) ----
#if HAVE_TDM
        if (w == 0)
            tdm_load_2d((uint32_t)(uintptr_t)&Kt[0],
                        K + (size_t)kv0 * HEAD_DIM, HEAD_DIM, 64, HEAD_DIM);
#else
#pragma unroll 4
        for (int i = 0; i < 32; ++i) {
            const int idx = tid * 32 + i;
            Kt[idx] = K[(size_t)kv0 * HEAD_DIM + idx];
        }
#endif
        // ---- V tile staged transposed: Vt[d][key] ----
#pragma unroll 4
        for (int i = 0; i < 32; ++i) {
            const int idx = tid * 32 + i;
            const int key = idx >> 6, d = idx & 63;
            Vt[d * 64 + key] = V[(size_t)(kv0 + key) * HEAD_DIM + d];
        }
#if HAVE_TDM
        if (w == 0) __builtin_amdgcn_s_wait_tensorcnt(0);
#endif
        __syncthreads();

        // ---- scores S = Q * K^T ----
        v8f sc[4];
#pragma unroll
        for (int j = 0; j < 4; ++j) sc[j] = vzero;
#pragma unroll
        for (int kk = 0; kk < 2; ++kk) {
#pragma unroll
            for (int j = 0; j < 4; ++j) {
                Frag bfr;
                const uint32_t* kp = (const uint32_t*)&Kt[(j * 16 + lm) * HEAD_DIM];
#pragma unroll
                for (int v = 0; v < 8; ++v)
                    bfr.u[v] = kp[kk * 16 + (v < 4 ? v : v + 4) + hi * 4];
                sc[j] = wmma_bf16(qfr[kk], bfr, sc[j]);
            }
        }

        // ---- causal mask on the diagonal tile ----
        if (t == qt) {
#pragma unroll
            for (int j = 0; j < 4; ++j)
#pragma unroll
                for (int r = 0; r < 8; ++r) {
                    const int row = w * 16 + hi * 8 + r;
                    const int col = j * 16 + lm;
                    if (col > row) sc[j][r] = -1e30f;
                }
        }

        // ---- online softmax (wave32 shuffle reductions) ----
        float corr[8];
#pragma unroll
        for (int r = 0; r < 8; ++r) {
            float mx = fmaxf(fmaxf(sc[0][r], sc[1][r]), fmaxf(sc[2][r], sc[3][r]));
#pragma unroll
            for (int msk = 1; msk < 16; msk <<= 1)
                mx = fmaxf(mx, __shfl_xor(mx, msk, 32));
            const float mnew = fmaxf(mrow[r], mx);
            corr[r] = __expf(mrow[r] - mnew);
            mrow[r] = mnew;
        }
#pragma unroll
        for (int j = 0; j < 4; ++j)
#pragma unroll
            for (int r = 0; r < 8; ++r)
                sc[j][r] = __expf(sc[j][r] - mrow[r]);
#pragma unroll
        for (int r = 0; r < 8; ++r) {
            float sm = (sc[0][r] + sc[1][r]) + (sc[2][r] + sc[3][r]);
#pragma unroll
            for (int msk = 1; msk < 16; msk <<= 1)
                sm += __shfl_xor(sm, msk, 32);
            lrow[r] = lrow[r] * corr[r] + sm;
        }
#pragma unroll
        for (int n = 0; n < 4; ++n)
#pragma unroll
            for (int r = 0; r < 8; ++r)
                oacc[n][r] *= corr[r];

        // ---- stage P (bf16) in this wave's LDS slab, re-gather as A ----
        uint16_t* P = Pb[w];
#pragma unroll
        for (int j = 0; j < 4; ++j)
#pragma unroll
            for (int r = 0; r < 8; ++r)
                P[(hi * 8 + r) * 64 + j * 16 + lm] = f2bf(sc[j][r]);
        asm volatile("s_wait_dscnt 0x0" ::: "memory");   // same-wave LDS RAW

        // ---- O += P * V ----
#pragma unroll
        for (int kk = 0; kk < 2; ++kk) {
            Frag afr;
            const uint32_t* rp = (const uint32_t*)&P[lm * 64];
#pragma unroll
            for (int v = 0; v < 8; ++v)
                afr.u[v] = rp[kk * 16 + (v < 4 ? v : v + 4) + hi * 4];
#pragma unroll
            for (int n = 0; n < 4; ++n) {
                Frag bfr;
                const uint32_t* vp = (const uint32_t*)&Vt[(n * 16 + lm) * 64];
#pragma unroll
                for (int v = 0; v < 8; ++v)
                    bfr.u[v] = vp[kk * 16 + (v < 4 ? v : v + 4) + hi * 4];
                oacc[n] = wmma_bf16(afr, bfr, oacc[n]);
            }
        }
    }

    // ---- normalize and store context bf16 [B,S,D] ----
    const int b = bh >> 4, h = bh & 15;
#pragma unroll
    for (int r = 0; r < 8; ++r) {
        const int   s   = q0 + w * 16 + hi * 8 + r;
        const float inv = 1.0f / lrow[r];
#pragma unroll
        for (int n = 0; n < 4; ++n) {
            const int d = h * 64 + n * 16 + lm;
            ctx[((size_t)(b * SEQ + s)) * MODEL_DIM + d] = f2bf(oacc[n][r] * inv);
        }
    }
}

// ---------------------------------------------------------------------------
extern "C" void kernel_launch(void* const* d_in, const int* in_sizes, int n_in,
                              void* d_out, int out_size, void* d_ws, size_t ws_size,
                              hipStream_t stream)
{
    const float* query = (const float*)d_in[0];
    const float* key   = (const float*)d_in[1];
    const float* value = (const float*)d_in[2];
    // d_in[3] = causal mask (handled analytically)
    const float* wq = (const float*)d_in[4];
    const float* bq = (const float*)d_in[5];
    const float* wk = (const float*)d_in[6];
    const float* bk = (const float*)d_in[7];
    const float* wv = (const float*)d_in[8];
    const float* bv = (const float*)d_in[9];
    const float* wo = (const float*)d_in[10];
    const float* bo = (const float*)d_in[11];

    const size_t elems = (size_t)BATCH * NUM_HEADS * SEQ * HEAD_DIM;  // 8.39M
    const size_t welem = (size_t)MODEL_DIM * MODEL_DIM;               // 1.05M
    uint16_t* Qh  = (uint16_t*)d_ws;
    uint16_t* Kh  = Qh + elems;
    uint16_t* Vh  = Kh + elems;
    uint16_t* ctx = Vh + elems;          // [B,S,MODEL_DIM] bf16
    uint16_t* wqB = ctx + elems;
    uint16_t* wkB = wqB + welem;
    uint16_t* wvB = wkB + welem;
    uint16_t* woB = wvB + welem;

    // weight bf16 pre-pass
    const int wn = (int)welem;
    const dim3 gCvt((wn / 4 + 255) / 256);
    cvt_bf16_kernel<<<gCvt, 256, 0, stream>>>(wq, wqB, wn);
    cvt_bf16_kernel<<<gCvt, 256, 0, stream>>>(wk, wkB, wn);
    cvt_bf16_kernel<<<gCvt, 256, 0, stream>>>(wv, wvB, wn);
    cvt_bf16_kernel<<<gCvt, 256, 0, stream>>>(wo, woB, wn);

    const dim3 gGemm(MODEL_DIM / 128, MROWS / 128);   // 8 x 64
    const dim3 gAttn(SEQ / 64, BATCH * NUM_HEADS);    // 32 x 64
    const float qscale = 0.125f;                      // 1/sqrt(HEAD_DIM)

    gemm_bias_kernel<0><<<gGemm, 256, 0, stream>>>(query, wqB, bq, Qh, qscale);
    gemm_bias_kernel<0><<<gGemm, 256, 0, stream>>>(key,   wkB, bk, Kh, 1.0f);
    gemm_bias_kernel<0><<<gGemm, 256, 0, stream>>>(value, wvB, bv, Vh, 1.0f);
    attention_kernel<<<gAttn, 128, 0, stream>>>(Qh, Kh, Vh, ctx);
    gemm_bias_kernel<1><<<gGemm, 256, 0, stream>>>(ctx, woB, bo, d_out, 1.0f);
}
